// CausalSelfAttention_16810501996829
// MI455X (gfx1250) — matrix operand order repaired and checked by reading, not compiled, not run on hardware
//
#include <hip/hip_runtime.h>

// ---------------------------------------------------------------------------
// CDNA5 (gfx1250) fused causal self-attention block, f16 WMMA + f32 softmax.
// Round 3: ping-pong register buffers (no rotation moves), 2x-unrolled
// steady-state GEMM loops, parity-split attention block loop.
// ---------------------------------------------------------------------------

typedef __attribute__((ext_vector_type(16))) _Float16 v16h;
typedef __attribute__((ext_vector_type(8)))  _Float16 v8h;
typedef __attribute__((ext_vector_type(8)))  float    v8f;

#define B_    2
#define T_    2048
#define D_    1024
#define H_    16
#define HD_   64
#define GIN_  12
#define EPS_     1.1920929e-07f
#define ASCALE_  0.1f
#define NEGLOG10K_ (-9.210340371976184f)   /* -ln(10000) */

// ----------------------------- device helpers ------------------------------

__device__ __forceinline__ v8f vzero8() {
  v8f v;
#pragma unroll
  for (int i = 0; i < 8; ++i) v[i] = 0.0f;
  return v;
}

__device__ __forceinline__ v8f wmma_f16(v16h a, v16h b, v8f c) {
  // v_wmma_f32_16x16x32_f16  D = A x B + C
  return __builtin_amdgcn_wmma_f32_16x16x32_f16(false, a, false, b, (short)0, c,
                                                false, false);
}

// A-fragment (16x32, f16) from a row-major [M][>=32] tile.
// lane 0-15  : row m=lane, K chunks {0..7, 16..23}
// lane 16-31 : row m=lane-16, K chunks {8..15, 24..31}
__device__ __forceinline__ v16h load_a_frag(const _Float16* tile, int ld, int lane) {
  const int m  = lane & 15;
  const int kh = (lane >> 4) << 3;            // 0 or 8
  const _Float16* p = tile + (size_t)m * ld + kh;
  const v8h lo = *(const v8h*)(p);
  const v8h hi = *(const v8h*)(p + 16);
  v16h a;
#pragma unroll
  for (int i = 0; i < 8; ++i) { a[i] = lo[i]; a[i + 8] = hi[i]; }
  return a;
}

// B-fragment (32x16, f16). B[k][n] taken from a row-major [N][K] array
// (column n of B == contiguous row n of the array). lane 0-15: n=lane,
// K chunks {0..7, 16..23} ; lane 16-31: n=lane-16, K chunks {8..15, 24..31}.
__device__ __forceinline__ v16h load_b_frag(const _Float16* tile, int ld, int lane) {
  const int n  = lane & 15;
  const int kb = (lane >> 4) << 3;            // 0 or 8
  const _Float16* p = tile + (size_t)n * ld + kb;
  const v8h lo = *(const v8h*)(p);
  const v8h hi = *(const v8h*)(p + 16);
  v16h b;
#pragma unroll
  for (int i = 0; i < 8; ++i) { b[i] = lo[i]; b[i + 8] = hi[i]; }
  return b;
}

// Reductions across the 16-lane half-groups of the WMMA C-layout (wave32).
__device__ __forceinline__ float red_sum16(float v) {
  v += __shfl_xor(v, 1, 32);
  v += __shfl_xor(v, 2, 32);
  v += __shfl_xor(v, 4, 32);
  v += __shfl_xor(v, 8, 32);
  return v;
}
__device__ __forceinline__ float red_max16(float v) {
  v = fmaxf(v, __shfl_xor(v, 1, 32));
  v = fmaxf(v, __shfl_xor(v, 2, 32));
  v = fmaxf(v, __shfl_xor(v, 4, 32));
  v = fmaxf(v, __shfl_xor(v, 8, 32));
  return v;
}

// ------------------------------- kernels -----------------------------------

__global__ void cvt_f32_f16_kernel(const float* __restrict__ in,
                                   _Float16* __restrict__ out, int n) {
  for (int i = blockIdx.x * blockDim.x + threadIdx.x; i < n;
       i += gridDim.x * blockDim.x)
    out[i] = (_Float16)in[i];
}

__global__ void gate_kernel(const float* __restrict__ x,
                            const float* __restrict__ Wg,
                            float* __restrict__ gate) {
  const int row = blockIdx.x * blockDim.x + threadIdx.x;
  if (row >= B_ * T_) return;
  float xv[GIN_];
#pragma unroll
  for (int i = 0; i < GIN_; ++i) xv[i] = x[(size_t)row * D_ + i];
#pragma unroll
  for (int h = 0; h < H_; ++h) {
    float z = 0.0f;
#pragma unroll
    for (int i = 0; i < GIN_; ++i) z += xv[i] * Wg[h * GIN_ + i];
    gate[(size_t)row * H_ + h] = 1.0f / (1.0f + __expf(-z));
  }
}

// One wave per (head, 32-row tile, {q,k,v}).  C tile = 32 rows x 64 (one head),
// ping-pong K loop: buffers A/B alternate with no register rotation.
// Epilogue: q/k -> RMS-norm + rotary -> [b][h][t][d] f16
//           v   -> blend with v1     -> [b][h][d][t] f16 (transposed for PV)
__global__ void __launch_bounds__(32)
qkv_kernel(const _Float16* __restrict__ x16,
           const _Float16* __restrict__ wq16,
           const _Float16* __restrict__ wk16,
           const _Float16* __restrict__ wv16,
           const float* __restrict__ v1,
           const float* __restrict__ lamb_p,
           _Float16* __restrict__ q16,
           _Float16* __restrict__ k16,
           _Float16* __restrict__ vT16) {
  const int lane  = threadIdx.x & 31;
  const int h     = blockIdx.x;                 // head
  const int r0    = blockIdx.y * 32;            // global row (b*T + t)
  const int which = blockIdx.z;                 // 0=q 1=k 2=v
  const _Float16* W = (which == 0) ? wq16 : (which == 1) ? wk16 : wv16;
  const int n0 = h * HD_;

  v8f acc[2][4];
#pragma unroll
  for (int rg = 0; rg < 2; ++rg)
#pragma unroll
    for (int f = 0; f < 4; ++f) acc[rg][f] = vzero8();

  const _Float16* A0 = x16 + (size_t)r0 * D_;
  const _Float16* A1 = A0 + (size_t)16 * D_;

  auto loadset = [&](int kk, v16h& a0, v16h& a1, v16h (&b)[4]) {
    a0 = load_a_frag(A0 + kk, D_, lane);
    a1 = load_a_frag(A1 + kk, D_, lane);
#pragma unroll
    for (int f = 0; f < 4; ++f)
      b[f] = load_b_frag(W + (size_t)(n0 + 16 * f) * D_ + kk, D_, lane);
  };
  auto compset = [&](const v16h& a0, const v16h& a1, const v16h (&b)[4]) {
#pragma unroll
    for (int f = 0; f < 4; ++f) acc[0][f] = wmma_f16(a0, b[f], acc[0][f]);
#pragma unroll
    for (int f = 0; f < 4; ++f) acc[1][f] = wmma_f16(a1, b[f], acc[1][f]);
  };

  v16h a0A, a1A, bA[4];
  v16h a0B, a1B, bB[4];
  loadset(0, a0A, a1A, bA);
  int k0 = 0;
  for (; k0 < D_ - 64; k0 += 64) {              // steady state, no rotation
    __builtin_prefetch(A0 + k0 + 256, 0, 1);
    loadset(k0 + 32, a0B, a1B, bB);
    compset(a0A, a1A, bA);
    loadset(k0 + 64, a0A, a1A, bA);
    compset(a0B, a1B, bB);
  }
  loadset(k0 + 32, a0B, a1B, bB);               // k0 == D_-64 here
  compset(a0A, a1A, bA);
  compset(a0B, a1B, bB);

  const int lg = lane >> 4, l16 = lane & 15;
  if (which == 2) {                             // ---- V: blend + transpose ----
    const float lam = *lamb_p;
#pragma unroll
    for (int rg = 0; rg < 2; ++rg) {
#pragma unroll
      for (int r = 0; r < 8; ++r) {
        const int m = rg * 16 + r + 8 * lg;
        const int grow = r0 + m;
        const int bb = grow >> 11;              // / T_
        const int t  = grow & (T_ - 1);
        const float* v1row = v1 + ((size_t)grow * H_ + h) * HD_;
        _Float16* col = vT16 + ((size_t)bb * H_ + h) * HD_ * T_ + t;
#pragma unroll
        for (int f = 0; f < 4; ++f) {
          const int n = 16 * f + l16;
          const float val = (1.0f - lam) * acc[rg][f][r] + lam * v1row[n];
          col[(size_t)n * T_] = (_Float16)val;
        }
      }
    }
  } else {                                      // ---- Q/K: RMS + RoPE ----
    float ss[2][8];
#pragma unroll
    for (int rg = 0; rg < 2; ++rg) {
#pragma unroll
      for (int r = 0; r < 8; ++r) {
        float s = 0.0f;
#pragma unroll
        for (int f = 0; f < 4; ++f) s += acc[rg][f][r] * acc[rg][f][r];
        ss[rg][r] = red_sum16(s);
      }
    }
    const float li  = NEGLOG10K_ / 32.0f;       // angle(d) = t * 10000^(-d/32)
    const float if0 = __expf(li * (float)l16);
    const float if1 = __expf(li * (float)(16 + l16));
    _Float16* dst = (which == 0) ? q16 : k16;
#pragma unroll
    for (int rg = 0; rg < 2; ++rg) {
#pragma unroll
      for (int r = 0; r < 8; ++r) {
        const int m = rg * 16 + r + 8 * lg;
        const int grow = r0 + m;
        const int bb = grow >> 11;
        const int t  = grow & (T_ - 1);
        const float sc = rsqrtf(ss[rg][r] * (1.0f / (float)HD_) + EPS_);
        float c0, s0, c1, s1;
        __sincosf((float)t * if0, &s0, &c0);
        __sincosf((float)t * if1, &s1, &c1);
        const float x0 = acc[rg][0][r] * sc, x1 = acc[rg][1][r] * sc;
        const float x2 = acc[rg][2][r] * sc, x3 = acc[rg][3][r] * sc;
        _Float16* row = dst + (((size_t)bb * H_ + h) * T_ + t) * HD_;
        row[l16]      = (_Float16)( x0 * c0 + x2 * s0);
        row[16 + l16] = (_Float16)( x1 * c1 + x3 * s1);
        row[32 + l16] = (_Float16)(-x0 * s0 + x2 * c0);
        row[48 + l16] = (_Float16)(-x1 * s1 + x3 * c1);
      }
    }
  }
}

// Flash-style causal attention: one wave per (b, h, 16-query tile); key blocks
// of 32.  K^T fragments ping-pong between two named buffers (parity-split
// driver, no rotation moves); V fragments and next-block K fragments are
// issued before the softmax VALU work so loads overlap exp/shuffles and the
// LDS P relayout.
__global__ void __launch_bounds__(32)
attn_kernel(const _Float16* __restrict__ q16,
            const _Float16* __restrict__ k16,
            const _Float16* __restrict__ vT16,
            const float* __restrict__ gate,
            _Float16* __restrict__ att16) {
  __shared__ __align__(16) _Float16 Plds[16 * 40];   // 16x32 P tile, padded

  const int lane = threadIdx.x & 31;
  const int l16  = lane & 15, lg = lane >> 4;
  const int h  = blockIdx.x;
  const int q0 = blockIdx.y * 16;
  const int bz = blockIdx.z;
  const size_t bh = (size_t)bz * H_ + h;

  const _Float16* Q  = q16 + (bh * T_ + q0) * HD_;
  const _Float16* Km = k16 + bh * T_ * HD_;
  const _Float16* Vt = vT16 + bh * HD_ * T_;

  const v16h qa0 = load_a_frag(Q, HD_, lane);        // d = 0..31
  const v16h qa1 = load_a_frag(Q + 32, HD_, lane);   // d = 32..63

  v8f o[4];
#pragma unroll
  for (int f = 0; f < 4; ++f) o[f] = vzero8();
  float mrow[8], lrow[8];
#pragma unroll
  for (int r = 0; r < 8; ++r) { mrow[r] = -1e30f; lrow[r] = 0.0f; }

  auto loadK = [&](int j0, v16h (&kb)[4]) {
    const _Float16* Kj = Km + (size_t)j0 * HD_;
    kb[0] = load_b_frag(Kj,                 HD_, lane);
    kb[1] = load_b_frag(Kj + 32,            HD_, lane);
    kb[2] = load_b_frag(Kj + 16 * HD_,      HD_, lane);
    kb[3] = load_b_frag(Kj + 16 * HD_ + 32, HD_, lane);
  };

  auto do_block = [&](int j0, const v16h (&kcur)[4], v16h (&knext)[4],
                      bool loadNext) {
    // S[16][32] = Q x K^T
    v8f s0 = vzero8(), s1 = vzero8();
    s0 = wmma_f16(qa0, kcur[0], s0);
    s0 = wmma_f16(qa1, kcur[1], s0);
    s1 = wmma_f16(qa0, kcur[2], s1);
    s1 = wmma_f16(qa1, kcur[3], s1);

    // early-issue: V fragments of this block, K^T fragments of next block
    v16h bv[4];
#pragma unroll
    for (int f = 0; f < 4; ++f)
      bv[f] = load_b_frag(Vt + (size_t)(16 * f) * T_ + j0, T_, lane);
    if (loadNext) loadK(j0 + 32, knext);

    // scale + causal mask, online softmax (overlaps the loads above)
    const int jj0 = j0 + l16, jj1 = j0 + 16 + l16;
    float bm[8];
#pragma unroll
    for (int r = 0; r < 8; ++r) {
      const int qi = q0 + r + 8 * lg;
      const float a0 = (jj0 <= qi) ? s0[r] * ASCALE_ : -1e30f;
      const float a1 = (jj1 <= qi) ? s1[r] * ASCALE_ : -1e30f;
      s0[r] = a0; s1[r] = a1;
      bm[r] = red_max16(fmaxf(a0, a1));
    }
    float alpha[8];
#pragma unroll
    for (int r = 0; r < 8; ++r) {
      const float mn = fmaxf(mrow[r], bm[r]);
      alpha[r] = __expf(mrow[r] - mn);
      mrow[r]  = mn;
      const float p0 = __expf(s0[r] - mn);
      const float p1 = __expf(s1[r] - mn);
      s0[r] = p0; s1[r] = p1;
      lrow[r] = lrow[r] * alpha[r] + red_sum16(p0 + p1);
    }
#pragma unroll
    for (int f = 0; f < 4; ++f) {
#pragma unroll
      for (int r = 0; r < 8; ++r) o[f][r] *= alpha[r];
    }

    __syncthreads();                          // LDS reuse fence
#pragma unroll
    for (int r = 0; r < 8; ++r) {
      const int m = r + 8 * lg;
      Plds[m * 40 + l16]      = (_Float16)s0[r];
      Plds[m * 40 + 16 + l16] = (_Float16)s1[r];
    }
    __syncthreads();

    const v16h pa = load_a_frag(Plds, 40, lane);   // P as A-fragment (K=32 keys)
#pragma unroll
    for (int f = 0; f < 4; ++f) o[f] = wmma_f16(pa, bv[f], o[f]);
  };

  const int nblk = (q0 + 16 + 31) / 32;
  const int jend = nblk * 32;
  v16h kA[4], kB[4];
  loadK(0, kA);
  if (nblk & 1) {                       // odd count: peel one block (kA)
    do_block(0, kA, kB, nblk > 1);
    for (int j0 = 32; j0 < jend; j0 += 64) {
      do_block(j0, kB, kA, true);
      do_block(j0 + 32, kA, kB, j0 + 64 < jend);
    }
  } else {                              // even count: pairs starting at kA
    for (int j0 = 0; j0 < jend; j0 += 64) {
      do_block(j0, kA, kB, true);
      do_block(j0 + 32, kB, kA, j0 + 64 < jend);
    }
  }

  // epilogue: 1/l, gate, store f16 -> att[(b*T+t)][h*64+d]
#pragma unroll
  for (int r = 0; r < 8; ++r) {
    const int m = r + 8 * lg;
    const size_t grow = (size_t)bz * T_ + q0 + m;
    const float g   = gate[grow * H_ + h];
    const float inv = g / lrow[r];
    _Float16* dst = att16 + grow * D_ + h * HD_;
    dst[l16]      = (_Float16)(o[0][r] * inv);
    dst[16 + l16] = (_Float16)(o[1][r] * inv);
    dst[32 + l16] = (_Float16)(o[2][r] * inv);
    dst[48 + l16] = (_Float16)(o[3][r] * inv);
  }
}

// out = att16 @ Wproj^T, fp32 output.  32x64 tile per wave, ping-pong buffers.
__global__ void __launch_bounds__(32)
proj_kernel(const _Float16* __restrict__ att16,
            const _Float16* __restrict__ wp16,
            float* __restrict__ out) {
  const int lane = threadIdx.x & 31;
  const int n0 = blockIdx.x * 64;
  const int r0 = blockIdx.y * 32;

  v8f acc[2][4];
#pragma unroll
  for (int rg = 0; rg < 2; ++rg)
#pragma unroll
    for (int f = 0; f < 4; ++f) acc[rg][f] = vzero8();

  const _Float16* A0 = att16 + (size_t)r0 * D_;
  const _Float16* A1 = A0 + (size_t)16 * D_;

  auto loadset = [&](int kk, v16h& a0, v16h& a1, v16h (&b)[4]) {
    a0 = load_a_frag(A0 + kk, D_, lane);
    a1 = load_a_frag(A1 + kk, D_, lane);
#pragma unroll
    for (int f = 0; f < 4; ++f)
      b[f] = load_b_frag(wp16 + (size_t)(n0 + 16 * f) * D_ + kk, D_, lane);
  };
  auto compset = [&](const v16h& a0, const v16h& a1, const v16h (&b)[4]) {
#pragma unroll
    for (int f = 0; f < 4; ++f) acc[0][f] = wmma_f16(a0, b[f], acc[0][f]);
#pragma unroll
    for (int f = 0; f < 4; ++f) acc[1][f] = wmma_f16(a1, b[f], acc[1][f]);
  };

  v16h a0A, a1A, bA[4];
  v16h a0B, a1B, bB[4];
  loadset(0, a0A, a1A, bA);
  int k0 = 0;
  for (; k0 < D_ - 64; k0 += 64) {
    __builtin_prefetch(A0 + k0 + 256, 0, 1);
    loadset(k0 + 32, a0B, a1B, bB);
    compset(a0A, a1A, bA);
    loadset(k0 + 64, a0A, a1A, bA);
    compset(a0B, a1B, bB);
  }
  loadset(k0 + 32, a0B, a1B, bB);
  compset(a0A, a1A, bA);
  compset(a0B, a1B, bB);

  const int lg = lane >> 4, l16 = lane & 15;
#pragma unroll
  for (int rg = 0; rg < 2; ++rg) {
#pragma unroll
    for (int r = 0; r < 8; ++r) {
      const size_t grow = (size_t)r0 + rg * 16 + r + 8 * lg;
#pragma unroll
      for (int f = 0; f < 4; ++f)
        out[grow * D_ + n0 + 16 * f + l16] = acc[rg][f][r];
    }
  }
}

// ------------------------------ host launch --------------------------------

extern "C" void kernel_launch(void* const* d_in, const int* in_sizes, int n_in,
                              void* d_out, int out_size, void* d_ws, size_t ws_size,
                              hipStream_t stream) {
  (void)in_sizes; (void)n_in; (void)out_size; (void)ws_size;
  const float* x     = (const float*)d_in[0];   // (B,T,D)
  const float* v1    = (const float*)d_in[1];   // (B,T,H,HD)
  const float* Wq    = (const float*)d_in[2];   // (D,D)
  const float* Wk    = (const float*)d_in[3];
  const float* Wv    = (const float*)d_in[4];
  const float* Wproj = (const float*)d_in[5];
  const float* Wg    = (const float*)d_in[6];   // (H,12)
  const float* lamb  = (const float*)d_in[7];   // scalar
  float* out = (float*)d_out;

  // workspace layout (byte offsets)
  char* wsb = (char*)d_ws;
  const size_t XN = (size_t)B_ * T_ * D_;       // 4,194,304 elems
  const size_t WN = (size_t)D_ * D_;            // 1,048,576 elems
  size_t off = 0;
  _Float16* x16  = (_Float16*)(wsb + off); off += XN * 2;
  _Float16* wq16 = (_Float16*)(wsb + off); off += WN * 2;
  _Float16* wk16 = (_Float16*)(wsb + off); off += WN * 2;
  _Float16* wv16 = (_Float16*)(wsb + off); off += WN * 2;
  _Float16* wp16 = (_Float16*)(wsb + off); off += WN * 2;
  _Float16* q16  = (_Float16*)(wsb + off); off += XN * 2;
  _Float16* k16  = (_Float16*)(wsb + off); off += XN * 2;
  _Float16* vT16 = (_Float16*)(wsb + off); off += XN * 2;
  _Float16* att16= (_Float16*)(wsb + off); off += XN * 2;
  float*    gate = (float*)(wsb + off);    off += (size_t)B_ * T_ * H_ * 4;

  // 1) fp32 -> f16 conversions
  cvt_f32_f16_kernel<<<2048, 256, 0, stream>>>(x, x16, (int)XN);
  cvt_f32_f16_kernel<<<1024, 256, 0, stream>>>(Wq, wq16, (int)WN);
  cvt_f32_f16_kernel<<<1024, 256, 0, stream>>>(Wk, wk16, (int)WN);
  cvt_f32_f16_kernel<<<1024, 256, 0, stream>>>(Wv, wv16, (int)WN);
  cvt_f32_f16_kernel<<<1024, 256, 0, stream>>>(Wproj, wp16, (int)WN);

  // 2) gates (fp32, tiny)
  gate_kernel<<<(B_ * T_ + 255) / 256, 256, 0, stream>>>(x, Wg, gate);

  // 3) QKV projections + RMS/RoPE/v-blend epilogues (32-row tiles)
  qkv_kernel<<<dim3(H_, (B_ * T_) / 32, 3), 32, 0, stream>>>(
      x16, wq16, wk16, wv16, v1, lamb, q16, k16, vT16);

  // 4) flash attention per (b, h, 16-query tile)
  attn_kernel<<<dim3(H_, T_ / 16, B_), 32, 0, stream>>>(
      q16, k16, vT16, gate, att16);

  // 5) output projection (fp32 out)
  proj_kernel<<<dim3(D_ / 64, (B_ * T_) / 32), 32, 0, stream>>>(
      att16, wp16, out);
}
